// ImplicitWarpModule_28321014350053
// MI455X (gfx1250) — matrix operand order repaired and checked by reading, not compiled, not run on hardware
//
#include <hip/hip_runtime.h>

// ---------------------------------------------------------------------------
// MI455X (gfx1250) implementation of the implicit-warp attention module.
// Projections run on v_wmma_f32_16x16x32_bf16; everything else is f32 VALU.
// One wave (32 lanes) owns a 16-pixel tile. 8192 blocks x 32 threads.
// ---------------------------------------------------------------------------

typedef __attribute__((ext_vector_type(16))) __bf16        v16bf;
typedef __attribute__((ext_vector_type(8)))  float         v8f;
typedef __attribute__((ext_vector_type(4)))  unsigned int  u32x4;

union ABbits { u32x4 u[2]; v16bf v; };

__device__ __forceinline__ unsigned short f2bf(float f) {
  // round-to-nearest-even float -> bf16 bits
  unsigned u = __float_as_uint(f);
  u += 0x7FFFu + ((u >> 16) & 1u);
  return (unsigned short)(u >> 16);
}

// A fragment: 16x32 bf16 tile slice, ISA layout:
// lanes 0-15: row=lane,   K = {kb..kb+7, kb+16..kb+23} with kb=0
// lanes 16-31: row=lane-16, same with kb=8
__device__ __forceinline__ v16bf load_A_lds(const unsigned short* tile, int lane, int kt) {
  const int row = lane & 15;
  const int kb  = (lane >> 4) * 8;
  const unsigned short* p = tile + row * 128 + kt * 32 + kb;
  ABbits r;
  r.u[0] = *(const u32x4*)(p);
  r.u[1] = *(const u32x4*)(p + 16);   // +16 elements = K+16
  return r.v;
}

// B fragment: 32x16 bf16 = w^T slice. Lane holds output column n = nt*16+(lane&15),
// K = {kb..kb+7, kb+16..kb+23} with kb = kt*32 + (lane>>4)*8. w is row-major [64][128].
__device__ __forceinline__ v16bf load_B_glb(const unsigned short* w, int lane, int nt, int kt) {
  const int n  = nt * 16 + (lane & 15);
  const int kb = kt * 32 + (lane >> 4) * 8;
  const unsigned short* p = w + n * 128 + kb;
  ABbits r;
  r.u[0] = *(const u32x4*)(p);
  r.u[1] = *(const u32x4*)(p + 16);
  return r.v;
}

__device__ __forceinline__ v8f wmma_bf16(v16bf a, v16bf b, v8f c) {
  return __builtin_amdgcn_wmma_f32_16x16x32_bf16(false, a, false, b, (short)0, c, false, false);
}

// D (16x16 f32) layout: lane L, VGPR r -> D[M = (L>>4)*8 + r][N = L&15]
__device__ __forceinline__ void store_D(float* buf, int mrow0, int nt, v8f acc,
                                        const float* __restrict__ bias, int lane) {
  const int ncol = nt * 16 + (lane & 15);
  const int mb   = mrow0 + (lane >> 4) * 8;
  const float bv = bias[ncol];
#pragma unroll
  for (int r = 0; r < 8; ++r)
    buf[(mb + r) * 64 + ncol] = acc[r] + bv;
}

__global__ void convert_weights_kernel(const float* __restrict__ qw,
                                       const float* __restrict__ kw,
                                       const float* __restrict__ vw,
                                       unsigned short* __restrict__ dst) {
  int i = blockIdx.x * 256 + threadIdx.x;
  if (i >= 3 * 8192) return;
  const float* src = (i < 8192) ? qw : ((i < 16384) ? kw : vw);
  dst[i] = f2bf(src[i & 8191]);
}

__global__ __launch_bounds__(32)
void implicit_warp_kernel(const float* __restrict__ feat_supp,
                          const float* __restrict__ flow,
                          const float* __restrict__ feat_curr,
                          const float* __restrict__ q_b,
                          const float* __restrict__ k_b,
                          const float* __restrict__ v_b,
                          const unsigned short* __restrict__ w_bf16, // [qw|kw|vw] bf16
                          float* __restrict__ out) {
  __shared__ float s_invdim[32];                       // 10000^(-j/32)
  __shared__ float s_pew[512];                         // window PE, 4 x 128
  __shared__ __align__(16) unsigned short s_fcq[16 * 128]; // fc bf16 -> q f32 16x64
  __shared__ __align__(16) unsigned short s_fw[64 * 128];  // fw bf16 -> v f32 64x64
  __shared__ __align__(16) float s_k[64 * 64];             // k f32

  const int lane = threadIdx.x;
  const int half = lane >> 4;   // 0: PE-y / channels 0-31, 1: PE-x / channels 32-63
  const int p    = lane & 15;   // pixel row within tile

  // --- constant tables -----------------------------------------------------
  // log2(10000)/32 = 0.41524101186092029
  s_invdim[lane] = exp2f(-(float)lane * 0.41524101186092029f);
  __syncthreads();
  const float SCW = 6.28318530717958648f / 1.000001f;  // 2*pi / (card=1 + eps)
#pragma unroll 4
  for (int idx = lane; idx < 512; idx += 32) {
    int uv = idx >> 7, k = idx & 127;
    float coord = (k < 64) ? (float)(uv >> 1) : (float)(uv & 1);
    int j = (k & 63) >> 1;
    float ang = coord * SCW * s_invdim[j];
    s_pew[idx] = (k & 1) ? __cosf(ang) : __sinf(ang);
  }
  __syncthreads();

  // --- per-pixel flow / grid ----------------------------------------------
  const int gid = blockIdx.x * 16 + p;      // global pixel id over B*HW
  const int b   = gid >> 16;
  const int n   = gid & 65535;
  const int yy0 = n >> 8, xx0 = n & 255;
  const float fx = flow[(size_t)gid * 2 + 0];   // flow last dim is (x, y)
  const float fy = flow[(size_t)gid * 2 + 1];
  const float gfy = (float)yy0 + fy;
  const float gfx = (float)xx0 + fx;
  const float fly = floorf(gfy), flx = floorf(gfx);
  const int   iy = (int)fly,    ix = (int)flx;
  const float dy = gfy - fly,   dx = gfx - flx;

  // --- build fc tile (16 x 128 bf16) ---------------------------------------
  {
    const float* fc = feat_curr + (size_t)b * 64 * 65536 + (size_t)(32 * half) * 65536 + n;
    const float vang = (half ? dx : dy) * (6.28318530717958648f / 2.000001f);
    unsigned short* row = s_fcq + p * 128 + half * 64;
#pragma unroll 8
    for (int j = 0; j < 32; ++j) {
      float f = fc[(size_t)j * 65536];
      float s, c;
      __sincosf(vang * s_invdim[j], &s, &c);
      row[2 * j]     = f2bf(f + s);
      row[2 * j + 1] = f2bf(f + c);
    }
  }

  // --- build fw tile (64 x 128 bf16): gather + window PE -------------------
  {
    const float* fs = feat_supp + (size_t)b * 64 * 65536 + (size_t)(32 * half) * 65536;
#pragma unroll
    for (int uv = 0; uv < 4; ++uv) {
      int ycl = min(max(iy + (uv >> 1), 0), 255);
      int xcl = min(max(ix + (uv & 1), 0), 255);
      int lin = (ycl << 8) | xcl;
      unsigned short* row = s_fw + (p * 4 + uv) * 128 + half * 64;
      const float* pe = s_pew + uv * 128 + half * 64;
#pragma unroll 8
      for (int j = 0; j < 32; ++j) {
        float f = fs[(size_t)j * 65536 + lin];
        row[2 * j]     = f2bf(f + pe[2 * j]);
        row[2 * j + 1] = f2bf(f + pe[2 * j + 1]);
      }
    }
  }
  __syncthreads();

  const unsigned short* qw = w_bf16;
  const unsigned short* kw = w_bf16 + 64 * 128;
  const unsigned short* vw = w_bf16 + 2 * 64 * 128;

  // --- Q = fc @ qw^T (16x64); result overlays fc (A consumed into regs first)
  {
    v16bf a0 = load_A_lds(s_fcq, lane, 0);
    v16bf a1 = load_A_lds(s_fcq, lane, 1);
    v16bf a2 = load_A_lds(s_fcq, lane, 2);
    v16bf a3 = load_A_lds(s_fcq, lane, 3);
#pragma unroll
    for (int nt = 0; nt < 4; ++nt) {
      v8f acc = {0.f, 0.f, 0.f, 0.f, 0.f, 0.f, 0.f, 0.f};
      acc = wmma_bf16(a0, load_B_glb(qw, lane, nt, 0), acc);
      acc = wmma_bf16(a1, load_B_glb(qw, lane, nt, 1), acc);
      acc = wmma_bf16(a2, load_B_glb(qw, lane, nt, 2), acc);
      acc = wmma_bf16(a3, load_B_glb(qw, lane, nt, 3), acc);
      store_D((float*)s_fcq, 0, nt, acc, q_b, lane);
    }
  }

  // --- K = fw @ kw^T (64x64) into s_k --------------------------------------
#pragma unroll
  for (int mt = 0; mt < 4; ++mt) {
    v16bf a0 = load_A_lds(s_fw + mt * 16 * 128, lane, 0);
    v16bf a1 = load_A_lds(s_fw + mt * 16 * 128, lane, 1);
    v16bf a2 = load_A_lds(s_fw + mt * 16 * 128, lane, 2);
    v16bf a3 = load_A_lds(s_fw + mt * 16 * 128, lane, 3);
#pragma unroll
    for (int nt = 0; nt < 4; ++nt) {
      v8f acc = {0.f, 0.f, 0.f, 0.f, 0.f, 0.f, 0.f, 0.f};
      acc = wmma_bf16(a0, load_B_glb(kw, lane, nt, 0), acc);
      acc = wmma_bf16(a1, load_B_glb(kw, lane, nt, 1), acc);
      acc = wmma_bf16(a2, load_B_glb(kw, lane, nt, 2), acc);
      acc = wmma_bf16(a3, load_B_glb(kw, lane, nt, 3), acc);
      store_D(s_k, mt * 16, nt, acc, k_b, lane);
    }
  }

  // --- V = fw @ vw^T (64x64); f32 rows exactly overlay the bf16 fw rows ----
  // (256B each); each mt's fw rows are consumed into A regs before overwrite.
#pragma unroll
  for (int mt = 0; mt < 4; ++mt) {
    v16bf a0 = load_A_lds(s_fw + mt * 16 * 128, lane, 0);
    v16bf a1 = load_A_lds(s_fw + mt * 16 * 128, lane, 1);
    v16bf a2 = load_A_lds(s_fw + mt * 16 * 128, lane, 2);
    v16bf a3 = load_A_lds(s_fw + mt * 16 * 128, lane, 3);
#pragma unroll
    for (int nt = 0; nt < 4; ++nt) {
      v8f acc = {0.f, 0.f, 0.f, 0.f, 0.f, 0.f, 0.f, 0.f};
      acc = wmma_bf16(a0, load_B_glb(vw, lane, nt, 0), acc);
      acc = wmma_bf16(a1, load_B_glb(vw, lane, nt, 1), acc);
      acc = wmma_bf16(a2, load_B_glb(vw, lane, nt, 2), acc);
      acc = wmma_bf16(a3, load_B_glb(vw, lane, nt, 3), acc);
      store_D((float*)s_fw, mt * 16, nt, acc, v_b, lane);
    }
  }
  __syncthreads();

  // --- per-pixel 8-head 1x4 softmax attention ------------------------------
  const float* qv = (const float*)s_fcq;   // 16 x 64
  const float* kv = s_k;                   // 64 x 64
  const float* vv = (const float*)s_fw;    // 64 x 64
  float* outp = out + (size_t)b * 64 * 65536 + n;
  const float scale = 0.35355339059327373f;  // hd^-0.5
#pragma unroll
  for (int t = 0; t < 4; ++t) {
    const int h = half * 4 + t;
    const float* qp = qv + p * 64 + h * 8;
    float l[4];
#pragma unroll
    for (int uv = 0; uv < 4; ++uv) {
      const float* kp = kv + (p * 4 + uv) * 64 + h * 8;
      float d = 0.f;
#pragma unroll
      for (int e = 0; e < 8; ++e) d += qp[e] * kp[e];
      l[uv] = d * scale;
    }
    float m  = fmaxf(fmaxf(l[0], l[1]), fmaxf(l[2], l[3]));
    float e0 = __expf(l[0] - m), e1 = __expf(l[1] - m);
    float e2 = __expf(l[2] - m), e3 = __expf(l[3] - m);
    float inv = 1.f / (e0 + e1 + e2 + e3);
#pragma unroll
    for (int d = 0; d < 8; ++d) {
      const float* vp = vv + p * 4 * 64 + h * 8 + d;
      float o = e0 * vp[0] + e1 * vp[64] + e2 * vp[128] + e3 * vp[192];
      outp[(size_t)(h * 8 + d) * 65536] = o * inv;
    }
  }
}

extern "C" void kernel_launch(void* const* d_in, const int* in_sizes, int n_in,
                              void* d_out, int out_size, void* d_ws, size_t ws_size,
                              hipStream_t stream) {
  (void)in_sizes; (void)n_in; (void)out_size; (void)ws_size;
  const float* feat_supp = (const float*)d_in[0];   // (2,64,256,256)
  const float* flow      = (const float*)d_in[1];   // (2,256,256,2)
  const float* feat_curr = (const float*)d_in[2];   // (2,64,256,256)
  const float* q_w = (const float*)d_in[3];
  const float* q_b = (const float*)d_in[4];
  const float* k_w = (const float*)d_in[5];
  const float* k_b = (const float*)d_in[6];
  const float* v_w = (const float*)d_in[7];
  const float* v_b = (const float*)d_in[8];

  unsigned short* w_bf16 = (unsigned short*)d_ws;   // 3 * 64*128 bf16 = 48KB

  convert_weights_kernel<<<96, 256, 0, stream>>>(q_w, k_w, v_w, w_bf16);

  // 2*65536 pixels / 16 per wave = 8192 blocks of one wave32 each
  implicit_warp_kernel<<<8192, 32, 0, stream>>>(
      feat_supp, flow, feat_curr, q_b, k_b, v_b, w_bf16, (float*)d_out);
}
